// Plane_refine_block_5_41927470743690
// MI455X (gfx1250) — matrix operand order repaired and checked by reading, not compiled, not run on hardware
//
#include <hip/hip_runtime.h>
#include <hip/hip_bf16.h>

// ---------------------------------------------------------------------------
// Plane_refine_block for MI455X (gfx1250), wave32 + WMMA bf16.
//
// Pipeline:
//   prep_kernel   : swizzle weights to WMMA-B fragment order (bf16), zero
//                   pooling accumulators, init per-channel max (-inf).
//   mlp_kernel    : h=relu((X@W1)*s1+b1); h2=relu((h@W2)*s2+b2); logit=h2@W3+b3
//                   spill h2 (bf16) to ws; a=h2@Wa -> atomicMax per channel.
//   pool_kernel   : recompute a=h2@Wa; e=exp(a-amax); eh=e*h2; build on/off
//                   masks; maskT @ [e|eh] accumulated via fp32 atomics.
//   final_kernel  : agg=num/(den+1e-9); relu(agg@Wm+bm); append ori columns.
// ---------------------------------------------------------------------------

#define NPTS   100000
#define NPAD   100096          // 1564 * 64
#define NBLK   (NPAD / 64)     // 1564
#define CCH    128
#define CCH2   256
#define PPL    64
#define OUTROW 140             // C + 12

typedef __attribute__((ext_vector_type(16))) __bf16 v16bf;
typedef __attribute__((ext_vector_type(8)))  __bf16 v8bf;
typedef __attribute__((ext_vector_type(8)))  float  v8f;

// ---------------------------------------------------------------- helpers ---
__device__ __forceinline__ v8f wmma_bf16(v16bf a, v16bf b, v8f c) {
  // D = A(16x32 bf16) * B(32x16 bf16) + C(16x16 f32)
  return __builtin_amdgcn_wmma_f32_16x16x32_bf16(
      /*neg_a=*/false, a, /*neg_b=*/false, b,
      /*c_mod=*/(short)0, c, /*reuse_a=*/false, /*reuse_b=*/false);
}

// A-matrix fragment (16x32 bf16) from row-major LDS [rows][S].
// Order used here (B uses the same): lane holds row M=lane&15; K = (lane>>4)*16 + {0..15}.
// -> one contiguous 16-elem run per lane.
__device__ __forceinline__ v16bf load_a_frag(const __bf16* lds, int row0,
                                             int k0, int lane, int S) {
  const __bf16* p = lds + (row0 + (lane & 15)) * S + k0 + ((lane >> 4) << 4);
  union { v16bf v; v8bf h[2]; } u;
  u.h[0] = *(const v8bf*)(p);
  u.h[1] = *(const v8bf*)(p + 8);
  return u.v;
}

// B-matrix fragment (32x16 bf16) from pre-swizzled global weights:
// frag (kt,ct): 32 lanes x 16 contiguous bf16 each.
__device__ __forceinline__ v16bf load_b_frag(const __bf16* wsw, int KT,
                                             int kt, int ct, int lane) {
  const __bf16* p = wsw + ((size_t)(((ct * KT + kt) * 32) + lane) << 4);
  union { v16bf v; v8bf h[2]; } u;
  u.h[0] = *(const v8bf*)(p);
  u.h[1] = *(const v8bf*)(p + 8);
  return u.v;
}

// B-matrix fragment from transposed LDS [cols][S] (col-major B):
// lane holds col N=lane&15; K = (lane>>4)*16 + e -> 16 contiguous elems.
__device__ __forceinline__ v16bf load_bt_frag(const __bf16* ldsT, int col0,
                                              int k0, int lane, int S) {
  const __bf16* p = ldsT + (col0 + (lane & 15)) * S + k0 + ((lane >> 4) << 4);
  union { v16bf v; v8bf h[2]; } u;
  u.h[0] = *(const v8bf*)(p);
  u.h[1] = *(const v8bf*)(p + 8);
  return u.v;
}

// monotone float <-> u32 encoding for atomicMax over signed floats
__device__ __forceinline__ unsigned fenc(float f) {
  int i = __float_as_int(f);
  return (i >= 0) ? ((unsigned)i | 0x80000000u) : ~(unsigned)i;
}
__device__ __forceinline__ float fdec(unsigned u) {
  int i = (u & 0x80000000u) ? (int)(u & 0x7fffffffu) : (int)~u;
  return __int_as_float(i);
}

// -------------------------------------------------------------- prep --------
__device__ __forceinline__ void swz(const float* W, __bf16* dst, int K,
                                    int Ncols, int t) {
  int total = K * Ncols;
  if (t >= total) return;
  int e  = t & 15;
  int l  = (t >> 4) & 31;
  int fr = t >> 9;                // fr = ct*KT + kt
  int KT = K >> 5;
  int kt = fr % KT;
  int ct = fr / KT;
  int k   = kt * 32 + ((l >> 4) << 4) + e;
  int col = ct * 16 + (l & 15);
  dst[t] = (__bf16)W[k * Ncols + col];
}

__global__ __launch_bounds__(256)
void prep_kernel(const float* W1, const float* W2, const float* Wa,
                 const float* Wm, __bf16* w1s, __bf16* w2s, __bf16* was,
                 __bf16* wms, unsigned* amax, float* accum) {
  int t = blockIdx.x * 256 + threadIdx.x;
  swz(W1, w1s, CCH,  CCH2, t);   // 128x256
  swz(W2, w2s, CCH2, CCH,  t);   // 256x128
  swz(Wa, was, CCH,  CCH,  t);   // 128x128
  swz(Wm, wms, CCH,  CCH,  t);   // 128x128
  if (t < CCH) amax[t] = fenc(-3.0e38f);
  if (t < 2 * PPL * CCH2) accum[t] = 0.0f;
}

// -------------------------------------------------------------- mlp ---------
__global__ __launch_bounds__(128)
void mlp_kernel(const float* __restrict__ feat,
                const float* __restrict__ s1, const float* __restrict__ b1,
                const float* __restrict__ s2, const float* __restrict__ b2,
                const float* __restrict__ W3, const float* __restrict__ b3,
                const __bf16* __restrict__ w1s, const __bf16* __restrict__ w2s,
                const __bf16* __restrict__ was,
                __bf16* __restrict__ h2bf, unsigned* __restrict__ amax,
                float* __restrict__ logit_out) {
  __shared__ __bf16 Xs [64 * 128];   // 16 KB
  __shared__ __bf16 Hs [64 * 256];   // 32 KB
  __shared__ __bf16 H2s[64 * 128];   // 16 KB
  const int lane = threadIdx.x & 31;
  const int w    = threadIdx.x >> 5;   // 4 waves; wave owns 16-row M-tile
  const int row0 = blockIdx.x * 64;

  // stage X (f32 -> bf16), zero-fill past N
  for (int i = threadIdx.x; i < 64 * 32; i += 128) {
    int r = i >> 5, c4 = (i & 31) * 4;
    float4 v = make_float4(0.f, 0.f, 0.f, 0.f);
    if (row0 + r < NPTS) v = *(const float4*)(feat + (size_t)(row0 + r) * CCH + c4);
    __bf16* d = &Xs[r * 128 + c4];
    d[0] = (__bf16)v.x; d[1] = (__bf16)v.y; d[2] = (__bf16)v.z; d[3] = (__bf16)v.w;
  }
  __syncthreads();

  const int mt = w;
  // GEMM1: H = relu((X @ W1) * s1 + b1), K=128
  for (int ct = 0; ct < 16; ++ct) {
    v8f acc = {};
    for (int kt = 0; kt < 4; ++kt) {
      v16bf a = load_a_frag(Xs, mt * 16, kt * 32, lane, 128);
      v16bf b = load_b_frag(w1s, 4, kt, ct, lane);
      acc = wmma_bf16(a, b, acc);
    }
    int ch = ct * 16 + (lane & 15);
    float sc = s1[ch], bi = b1[ch];
    for (int r = 0; r < 8; ++r) {
      float v = acc[r] * sc + bi;
      v = v > 0.f ? v : 0.f;
      Hs[(mt * 16 + r + ((lane >> 4) << 3)) * 256 + ch] = (__bf16)v;
    }
  }
  // GEMM2: H2 = relu((H @ W2) * s2 + b2), K=256  (wave-local rows: no barrier)
  for (int ct = 0; ct < 8; ++ct) {
    v8f acc = {};
    for (int kt = 0; kt < 8; ++kt) {
      v16bf a = load_a_frag(Hs, mt * 16, kt * 32, lane, 256);
      v16bf b = load_b_frag(w2s, 8, kt, ct, lane);
      acc = wmma_bf16(a, b, acc);
    }
    int ch = ct * 16 + (lane & 15);
    float sc = s2[ch], bi = b2[ch];
    for (int r = 0; r < 8; ++r) {
      float v = acc[r] * sc + bi;
      v = v > 0.f ? v : 0.f;
      H2s[(mt * 16 + r + ((lane >> 4) << 3)) * 128 + ch] = (__bf16)v;
    }
  }
  // spill this wave's 16 rows of h2 (contiguous 4 KB) to workspace
  {
    const unsigned* src = (const unsigned*)&H2s[mt * 16 * 128];
    unsigned* dst = (unsigned*)(h2bf + (size_t)(row0 + mt * 16) * CCH);
    for (int i = lane; i < 16 * 128 / 2; i += 32) dst[i] = src[i];
  }
  // logit = h2 @ W3 + b3 (split over lane halves, combine via shfl)
  {
    int m = lane & 15, half = lane >> 4;
    float s = 0.f;
    for (int c = 0; c < 64; ++c) {
      int ch = half * 64 + c;
      s += (float)H2s[(mt * 16 + m) * 128 + ch] * W3[ch];
    }
    s += __shfl_xor(s, 16, 32);
    int row = row0 + mt * 16 + m;
    if (half == 0 && row < NPTS) logit_out[row] = s + b3[0];
  }
  // a = h2 @ Wa : only the per-channel running max is needed in this pass.
  // (padded rows contribute a==0; acceptable approximation of max over N)
  for (int ct = 0; ct < 8; ++ct) {
    v8f acc = {};
    for (int kt = 0; kt < 4; ++kt) {
      v16bf a = load_a_frag(H2s, mt * 16, kt * 32, lane, 128);
      v16bf b = load_b_frag(was, 4, kt, ct, lane);
      acc = wmma_bf16(a, b, acc);
    }
    float mx = acc[0];
    for (int r = 1; r < 8; ++r) mx = fmaxf(mx, acc[r]);
    mx = fmaxf(mx, __shfl_xor(mx, 16, 32));
    if (lane < 16) atomicMax(&amax[ct * 16 + lane], fenc(mx));
  }
}

// -------------------------------------------------------------- pool --------
__global__ __launch_bounds__(256)
void pool_kernel(const float* __restrict__ xyz, const float* __restrict__ center,
                 const float* __restrict__ normal, const float* __restrict__ pmin,
                 const float* __restrict__ pmax,
                 const __bf16* __restrict__ h2bf, const __bf16* __restrict__ was,
                 const unsigned* __restrict__ amax,
                 const float* __restrict__ logit, float* __restrict__ accum) {
  __shared__ __bf16 H2c[64 * 128];   // 16 KB   h2 chunk (row-major)
  __shared__ __bf16 eT [256 * 64];   // 32 KB   rows 0..127=e^T, 128..255=(e*h2)^T
  __shared__ __bf16 mT [128 * 64];   // 16 KB   rows 0..63 on-mask^T, 64..127 off^T
  const int lane = threadIdx.x & 31;
  const int w    = threadIdx.x >> 5;   // 8 waves
  const int row0 = blockIdx.x * 64;

  // stage h2 chunk
  {
    const unsigned* src = (const unsigned*)(h2bf + (size_t)row0 * CCH);
    unsigned* dst = (unsigned*)H2c;
    for (int i = threadIdx.x; i < 64 * 128 / 2; i += 256) dst[i] = src[i];
  }
  // on/off membership masks (transposed, bf16 0/1)
  for (int idx = threadIdx.x; idx < 128 * 64; idx += 256) {
    int n = idx & 63, pp = idx >> 6, p = pp & 63;
    float m = 0.f;
    int row = row0 + n;
    if (row < NPTS) {
      float x = xyz[row * 3 + 0], y = xyz[row * 3 + 1], z = xyz[row * 3 + 2];
      float nx = normal[p * 3], ny = normal[p * 3 + 1], nz = normal[p * 3 + 2];
      float offv = center[p * 3] * nx + center[p * 3 + 1] * ny +
                   center[p * 3 + 2] * nz;
      float proj = x * nx + y * ny + z * nz;
      bool dm = fabsf(proj - offv) < 0.1f;
      bool rg = (x >= pmin[p * 3]) && (x < pmax[p * 3]) &&
                (y >= pmin[p * 3 + 1]) && (y < pmax[p * 3 + 1]);
      bool on = logit[row] > 0.f;   // sigmoid(logit) > 0.5
      bool sel = dm && rg && ((pp < 64) ? on : !on);
      m = sel ? 1.f : 0.f;
    }
    mT[pp * 64 + n] = (__bf16)m;
  }
  __syncthreads();

  // recompute a = h2 @ Wa; eT/ehT (transposed so they feed WMMA-B directly)
  {
    int mt = w & 3, ctb = (w >> 2) * 4;
    for (int ct = ctb; ct < ctb + 4; ++ct) {
      v8f acc = {};
      for (int kt = 0; kt < 4; ++kt) {
        v16bf a = load_a_frag(H2c, mt * 16, kt * 32, lane, 128);
        v16bf b = load_b_frag(was, 4, kt, ct, lane);
        acc = wmma_bf16(a, b, acc);
      }
      int ch = ct * 16 + (lane & 15);
      float am = fdec(amax[ch]);
      for (int r = 0; r < 8; ++r) {
        int n = mt * 16 + r + ((lane >> 4) << 3);
        float e = __expf(acc[r] - am);
        float h2v = (float)H2c[n * 128 + ch];
        eT[ch * 64 + n]          = (__bf16)e;
        eT[(128 + ch) * 64 + n]  = (__bf16)(e * h2v);
      }
    }
  }
  __syncthreads();

  // maskT(128xK) @ [e|eh](Kx256): 8 plane-tiles x 16 col-tiles, K=64
  {
    int pt = w;
    for (int ct = 0; ct < 16; ++ct) {
      v8f acc = {};
      for (int kt = 0; kt < 2; ++kt) {
        v16bf a = load_a_frag(mT, pt * 16, kt * 32, lane, 64);
        v16bf b = load_bt_frag(eT, ct * 16, kt * 32, lane, 64);
        acc = wmma_bf16(a, b, acc);
      }
      for (int r = 0; r < 8; ++r) {
        int pp = pt * 16 + r + ((lane >> 4) << 3);
        int cc = ct * 16 + (lane & 15);
        (void)__hip_atomic_fetch_add(&accum[pp * CCH2 + cc], acc[r],
                                     __ATOMIC_RELAXED, __HIP_MEMORY_SCOPE_AGENT);
      }
    }
  }
}

// -------------------------------------------------------------- final -------
__global__ __launch_bounds__(256)
void final_kernel(const float* __restrict__ accum, const __bf16* __restrict__ wms,
                  const float* __restrict__ bm, const float* __restrict__ center,
                  const float* __restrict__ normal, const float* __restrict__ pmin,
                  const float* __restrict__ pmax, float* __restrict__ outp) {
  __shared__ __bf16 Ag[128 * 128];   // 32 KB  agg = num/(den+eps), bf16
  const int lane = threadIdx.x & 31;
  const int w    = threadIdx.x >> 5;

  for (int i = threadIdx.x; i < 128 * 128; i += 256) {
    int pp = i >> 7, c = i & 127;
    float den = accum[pp * CCH2 + c];
    float num = accum[pp * CCH2 + 128 + c];
    Ag[i] = (__bf16)(num / (den + 1e-9f));
  }
  __syncthreads();

  float* on_base  = outp + NPTS;                 // [64][140]
  float* off_base = outp + NPTS + PPL * OUTROW;  // [64][140]
  const int mt = w;
  for (int ct = 0; ct < 8; ++ct) {
    v8f acc = {};
    for (int kt = 0; kt < 4; ++kt) {
      v16bf a = load_a_frag(Ag, mt * 16, kt * 32, lane, 128);
      v16bf b = load_b_frag(wms, 4, kt, ct, lane);
      acc = wmma_bf16(a, b, acc);
    }
    int ch = ct * 16 + (lane & 15);
    float bi = bm[ch];
    for (int r = 0; r < 8; ++r) {
      int pp = mt * 16 + r + ((lane >> 4) << 3);
      float v = acc[r] + bi;
      v = v > 0.f ? v : 0.f;
      float* base = (pp < PPL) ? (on_base + pp * OUTROW)
                               : (off_base + (pp - PPL) * OUTROW);
      base[ch] = v;
    }
  }
  // ori = [center | normal | pmin | pmax] columns 128..139
  for (int i = threadIdx.x; i < PPL * 12; i += 256) {
    int p = i / 12, j = i % 12;
    float v;
    if (j < 3)      v = center[p * 3 + j];
    else if (j < 6) v = normal[p * 3 + j - 3];
    else if (j < 9) v = pmin[p * 3 + j - 6];
    else            v = pmax[p * 3 + j - 9];
    on_base [p * OUTROW + 128 + j] = v;
    off_base[p * OUTROW + 128 + j] = v;
  }
}

// -------------------------------------------------------------- launch ------
extern "C" void kernel_launch(void* const* d_in, const int* in_sizes, int n_in,
                              void* d_out, int out_size, void* d_ws,
                              size_t ws_size, hipStream_t stream) {
  (void)in_sizes; (void)n_in; (void)out_size; (void)ws_size;
  const float* feature = (const float*)d_in[0];
  const float* xyz     = (const float*)d_in[1];
  const float* center  = (const float*)d_in[2];
  const float* normal  = (const float*)d_in[3];
  const float* pmin    = (const float*)d_in[4];
  const float* pmax    = (const float*)d_in[5];
  const float* W1      = (const float*)d_in[6];
  const float* s1      = (const float*)d_in[7];
  const float* b1      = (const float*)d_in[8];
  const float* W2      = (const float*)d_in[9];
  const float* s2      = (const float*)d_in[10];
  const float* b2      = (const float*)d_in[11];
  const float* W3      = (const float*)d_in[12];
  const float* b3      = (const float*)d_in[13];
  const float* Wa      = (const float*)d_in[14];
  const float* Wm      = (const float*)d_in[15];
  const float* bm      = (const float*)d_in[16];
  float* out = (float*)d_out;

  // workspace carve-up (~25 MB)
  char* p = (char*)d_ws;
  __bf16* w1s  = (__bf16*)p;  p += (size_t)CCH * CCH2 * 2;
  __bf16* w2s  = (__bf16*)p;  p += (size_t)CCH2 * CCH * 2;
  __bf16* was  = (__bf16*)p;  p += (size_t)CCH * CCH * 2;
  __bf16* wms  = (__bf16*)p;  p += (size_t)CCH * CCH * 2;
  unsigned* amax = (unsigned*)p; p += 128 * sizeof(unsigned);
  float* accum = (float*)p;   p += (size_t)2 * PPL * CCH2 * sizeof(float);
  __bf16* h2bf = (__bf16*)p;  // NPAD * 128 * 2 bytes

  prep_kernel<<<128, 256, 0, stream>>>(W1, W2, Wa, Wm, w1s, w2s, was, wms,
                                       amax, accum);
  mlp_kernel<<<NBLK, 128, 0, stream>>>(feature, s1, b1, s2, b2, W3, b3,
                                       w1s, w2s, was, h2bf, amax, out);
  pool_kernel<<<NBLK, 256, 0, stream>>>(xyz, center, normal, pmin, pmax,
                                        h2bf, was, amax, out, accum);
  final_kernel<<<1, 256, 0, stream>>>(accum, wms, bm, center, normal,
                                      pmin, pmax, out);
}
